// GAT_17480516895404
// MI455X (gfx1250) — matrix operand order repaired and checked by reading, not compiled, not run on hardware
//
#include <hip/hip_runtime.h>
#include <hip/hip_bf16.h>

// ---------- types ----------
typedef __attribute__((ext_vector_type(16))) __bf16 v16bf;
typedef __attribute__((ext_vector_type(8)))  float  v8f;

struct bpack32 { uint4 lo, hi; };   // 16 bf16 = 32 bytes, matches v16bf

__device__ __forceinline__ unsigned short f2bfu(float f) {
    unsigned u = __builtin_bit_cast(unsigned, f);
    unsigned r = u + 0x7FFFu + ((u >> 16) & 1u);   // RNE
    return (unsigned short)(r >> 16);
}
__device__ __forceinline__ __bf16 f2bf(float f) {
    return __builtin_bit_cast(__bf16, f2bfu(f));
}

// monotonic float<->uint mapping for atomicMax on floats
__device__ __forceinline__ unsigned fkey(float f) {
    unsigned u = __builtin_bit_cast(unsigned, f);
    return (u & 0x80000000u) ? ~u : (u | 0x80000000u);
}
__device__ __forceinline__ float fdec(unsigned k) {
    unsigned u = (k & 0x80000000u) ? (k ^ 0x80000000u) : ~k;
    return __builtin_bit_cast(float, u);
}

// ---------- zero init ----------
__global__ void zero_f32(float* __restrict__ p, int n) {
    int i = blockIdx.x * blockDim.x + threadIdx.x;
    if (i < n) p[i] = 0.0f;
}

// ---------- pack B (KxN f32, row-major) into per-lane WMMA bf16 layout ----------
// Bp layout: [ntile][kstep][lane] x 32B ; lane holds b[t] = B[32*ks + 8*h + t][n0 + r] (t<8), B[32*ks + 16 + 8*h + t-8][n0 + r] (t>=8)
__global__ void pack_B(const float* __restrict__ B, uint4* __restrict__ Bp,
                       int K, int NN, int ksteps, int total)
{
    int idx = blockIdx.x * blockDim.x + threadIdx.x;
    if (idx >= total) return;
    int lane = idx & 31;
    int ks   = (idx >> 5) % ksteps;
    int nt   = idx / (32 * ksteps);
    int r = lane & 15, hh = lane >> 4;
    int col = nt * 16 + r;
    unsigned d[8];
    #pragma unroll
    for (int j = 0; j < 8; ++j) {
        int k0 = 32 * ks + 8 * hh + 2 * (j & 3) + 16 * (j >> 2);
        float v0 = (k0     < K) ? B[(long long)k0 * NN + col]       : 0.0f;
        float v1 = (k0 + 1 < K) ? B[(long long)(k0 + 1) * NN + col] : 0.0f;
        d[j] = (unsigned)f2bfu(v0) | ((unsigned)f2bfu(v1) << 16);
    }
    Bp[idx * 2 + 0] = make_uint4(d[0], d[1], d[2], d[3]);
    Bp[idx * 2 + 1] = make_uint4(d[4], d[5], d[6], d[7]);
}

__device__ __forceinline__ void put8(v16bf& a, int base, float4 x, float4 y) {
    a[base + 0] = f2bf(x.x); a[base + 1] = f2bf(x.y);
    a[base + 2] = f2bf(x.z); a[base + 3] = f2bf(x.w);
    a[base + 4] = f2bf(y.x); a[base + 5] = f2bf(y.y);
    a[base + 6] = f2bf(y.z); a[base + 7] = f2bf(y.w);
}

// ---------- WMMA GEMM:  D[M,NN] = act(A[M,K] @ B + bias) ----------
// compile-time K, NN; M must be a multiple of 16; one wave per 16x16 tile
template<int K, int NN, bool BIAS, bool RELU>
__global__ __launch_bounds__(32)
void wmma_gemm_bf16(const float* __restrict__ A, const bpack32* __restrict__ Bp,
                    const float* __restrict__ bias, float* __restrict__ D, int M)
{
    constexpr int KSTEPS = (K + 31) / 32;
    const int lane = threadIdx.x & 31;
    const int r    = lane & 15;
    const int hh   = lane >> 4;
    const int m0   = blockIdx.x * 16;
    const int nt   = blockIdx.y;
    const int bcol = nt * 16 + r;

    const float*   ap  = A + (size_t)(m0 + r) * K;
    const bpack32* bpp = Bp + ((size_t)nt * KSTEPS) * 32 + lane;

    v8f acc = {};
    #pragma unroll
    for (int ks = 0; ks < KSTEPS; ++ks) {
        const int k0 = 32 * ks;
        v16bf a;
        // run 1: K-range [k0 + 8h, k0 + 8h + 8)  (always fully valid: K % 16 == 0)
        float4 a0 = *(const float4*)(ap + k0 + 8 * hh);
        float4 a1 = *(const float4*)(ap + k0 + 8 * hh + 4);
        put8(a, 0, a0, a1);
        // run 2: K-range [k0 + 16 + 8h, +8)  (valid iff k0 + 32 <= K, compile-time)
        if (k0 + 32 <= K) {
            float4 a2 = *(const float4*)(ap + k0 + 16 + 8 * hh);
            float4 a3 = *(const float4*)(ap + k0 + 16 + 8 * hh + 4);
            put8(a, 8, a2, a3);
        } else {
            #pragma unroll
            for (int j = 8; j < 16; ++j) a[j] = f2bf(0.0f);
        }
        bpack32 bw = bpp[ks * 32];
        v16bf b = __builtin_bit_cast(v16bf, bw);
        acc = __builtin_amdgcn_wmma_f32_16x16x32_bf16(false, a, false, b,
                                                      (short)0, acc, false, false);
    }
    float bv = BIAS ? bias[bcol] : 0.0f;
    float* dp = D + (size_t)m0 * NN + bcol;
    #pragma unroll
    for (int v = 0; v < 8; ++v) {
        float val = acc[v] + bv;
        if (RELU) val = fmaxf(val, 0.0f);
        dp[(size_t)(v + 8 * hh) * NN] = val;
    }
}

// ---------- per-node attention dots ----------
template<int C>
__global__ void attn_dots(const float* __restrict__ h, const float* __restrict__ asrc,
                          const float* __restrict__ adst, float* __restrict__ as_,
                          float* __restrict__ ad_, int N)
{
    int n = blockIdx.x * blockDim.x + threadIdx.x;
    if (n >= N) return;
    const float4* hp = (const float4*)(h + (size_t)n * C);
    const float4* sp = (const float4*)asrc;
    const float4* dp = (const float4*)adst;
    float s = 0.f, d = 0.f;
    #pragma unroll
    for (int c = 0; c < C / 4; ++c) {
        float4 v = hp[c], a = sp[c], b = dp[c];
        s += v.x * a.x + v.y * a.y + v.z * a.z + v.w * a.w;
        d += v.x * b.x + v.y * b.y + v.z * b.z + v.w * b.w;
    }
    as_[n] = s; ad_[n] = d;
}

__device__ __forceinline__ void edge_sd(const int* src, const int* dst, int e, int E,
                                        int& s, int& d)
{
    if (e < E) { s = src[e]; d = dst[e]; }
    else       { s = e - E; d = e - E; }   // virtual self-loop
}

// ---------- pass 1: segment max of leaky_relu(as[s]+ad[d]) over dst ----------
__global__ void edge_max(const int* __restrict__ src, const int* __restrict__ dst,
                         const float* __restrict__ as_, const float* __restrict__ ad_,
                         unsigned* __restrict__ emaxu, int EE, int E)
{
    int e = blockIdx.x * blockDim.x + threadIdx.x;
    if (e >= EE) return;
    int s, d; edge_sd(src, dst, e, E, s, d);
    float ev = as_[s] + ad_[d];
    float le = ev > 0.f ? ev : 0.2f * ev;
    atomicMax(&emaxu[d], fkey(le));
}

// ---------- pass 2: e_exp + segment sum ----------
__global__ void edge_exp(const int* __restrict__ src, const int* __restrict__ dst,
                         const float* __restrict__ as_, const float* __restrict__ ad_,
                         const unsigned* __restrict__ emaxu, float* __restrict__ denom,
                         float* __restrict__ eexp, int EE, int E)
{
    int e = blockIdx.x * blockDim.x + threadIdx.x;
    if (e >= EE) return;
    int s, d; edge_sd(src, dst, e, E, s, d);
    float ev = as_[s] + ad_[d];
    float le = ev > 0.f ? ev : 0.2f * ev;
    float ex = __expf(le - fdec(emaxu[d]));
    eexp[e] = ex;
    atomicAdd(&denom[d], ex);
}

// ---------- pass 3: agg[d] += alpha * h[s]  (edge x 4-channel chunks) ----------
template<int C>
__global__ void edge_agg(const int* __restrict__ src, const int* __restrict__ dst,
                         const float* __restrict__ eexp, const float* __restrict__ denom,
                         const float* __restrict__ h, float* __restrict__ agg,
                         int EE, int E)
{
    constexpr int C4 = C / 4;
    int idx = blockIdx.x * blockDim.x + threadIdx.x;
    if (idx >= EE * C4) return;
    int e = idx / C4, g = idx % C4;
    int s, d; edge_sd(src, dst, e, E, s, d);
    float alpha = eexp[e] / (denom[d] + 1e-16f);
    float4 hv = *(const float4*)(h + (size_t)s * C + g * 4);
    float* ap = agg + (size_t)d * C + g * 4;
    atomicAdd(&ap[0], alpha * hv.x);
    atomicAdd(&ap[1], alpha * hv.y);
    atomicAdd(&ap[2], alpha * hv.z);
    atomicAdd(&ap[3], alpha * hv.w);
}

// ---------- bias (+ optional relu), in-place capable ----------
__global__ void bias_act(const float* __restrict__ in, const float* __restrict__ bias,
                         float* __restrict__ out, int N, int C, int relu)
{
    int idx = blockIdx.x * blockDim.x + threadIdx.x;
    if (idx >= N * C) return;
    float v = in[idx] + bias[idx % C];
    if (relu) v = fmaxf(v, 0.0f);
    out[idx] = v;
}

// ---------- in-place row softmax over 16 ----------
__global__ void softmax16(float* __restrict__ p, int N)
{
    int n = blockIdx.x * blockDim.x + threadIdx.x;
    if (n >= N) return;
    float* row = p + (size_t)n * 16;
    float v[16], m = -3.402823466e+38f;
    #pragma unroll
    for (int i = 0; i < 16; ++i) { v[i] = row[i]; m = fmaxf(m, v[i]); }
    float s = 0.f;
    #pragma unroll
    for (int i = 0; i < 16; ++i) { v[i] = __expf(v[i] - m); s += v[i]; }
    float inv = 1.0f / s;
    #pragma unroll
    for (int i = 0; i < 16; ++i) row[i] = v[i] * inv;
}

extern "C" void kernel_launch(void* const* d_in, const int* in_sizes, int n_in,
                              void* d_out, int out_size, void* d_ws, size_t ws_size,
                              hipStream_t stream)
{
    const float* x      = (const float*)d_in[0];
    const int*   ei     = (const int*)  d_in[1];
    const float* W1     = (const float*)d_in[2];
    const float* a_src1 = (const float*)d_in[3];
    const float* a_dst1 = (const float*)d_in[4];
    const float* b1     = (const float*)d_in[5];
    const float* W2     = (const float*)d_in[6];
    const float* a_src2 = (const float*)d_in[7];
    const float* a_dst2 = (const float*)d_in[8];
    const float* b2     = (const float*)d_in[9];
    const float* lin1W  = (const float*)d_in[10];
    const float* lin1b  = (const float*)d_in[11];
    const float* lin2W  = (const float*)d_in[12];
    const float* lin2b  = (const float*)d_in[13];

    const int N  = in_sizes[0] / 128;   // 100000 (multiple of 16)
    const int E  = in_sizes[1] / 2;     // 1600000
    const int EE = E + N;               // + self loops
    constexpr int Fin = 128, Hid = 64, Out = 16;
    const int* src = ei;
    const int* dst = ei + E;

    // workspace carve-out (256B aligned)
    char* ws = (char*)d_ws;
    size_t off = 0;
    auto carve = [&](size_t bytes) -> void* {
        void* p = ws + off;
        off = (off + bytes + 255) & ~(size_t)255;
        return p;
    };
    float*    h0    = (float*)   carve((size_t)N * Hid * 4); // gemm1 out; later lin1 out
    float*    agg1  = (float*)   carve((size_t)N * Hid * 4); // agg1 -> h1 (in place)
    float*    eexp  = (float*)   carve((size_t)EE * 4);
    float*    h2    = (float*)   carve((size_t)N * Out * 4);
    float*    agg2  = (float*)   carve((size_t)N * Out * 4); // agg2 -> out2 (in place)
    float*    asv   = (float*)   carve((size_t)N * 4);
    float*    adv   = (float*)   carve((size_t)N * 4);
    unsigned* emaxu = (unsigned*)carve((size_t)N * 4);
    float*    denom = (float*)   carve((size_t)N * 4);
    // packed weights (bf16, WMMA B-layout). sizes: ntiles*ksteps*32 lanes * 32B
    uint4* BpW1 = (uint4*)carve((size_t)(Hid/16) * 4 * 32 * 32); // K=128 -> 4 ksteps
    uint4* BpW2 = (uint4*)carve((size_t)(Out/16) * 2 * 32 * 32); // K=64  -> 2 ksteps
    uint4* BpL1 = (uint4*)carve((size_t)(Hid/16) * 1 * 32 * 32); // K=16  -> 1 kstep
    uint4* BpL2 = (uint4*)carve((size_t)(Out/16) * 2 * 32 * 32); // K=64  -> 2 ksteps
    float* outp = (float*)d_out;

    const int T = 256;
    auto blocks = [&](long long n) { return (unsigned)((n + T - 1) / T); };
    const int tilesM = N / 16;

    // ---- pack all weights ----
    {
        int t1 = (Hid/16) * 4 * 32, t2 = (Out/16) * 2 * 32,
            t3 = (Hid/16) * 1 * 32, t4 = (Out/16) * 2 * 32;
        pack_B<<<blocks(t1), T, 0, stream>>>(W1,    BpW1, Fin, Hid, 4, t1);
        pack_B<<<blocks(t2), T, 0, stream>>>(W2,    BpW2, Hid, Out, 2, t2);
        pack_B<<<blocks(t3), T, 0, stream>>>(lin1W, BpL1, Out, Hid, 1, t3);
        pack_B<<<blocks(t4), T, 0, stream>>>(lin2W, BpL2, Hid, Out, 2, t4);
    }

    // ---- layer 1 ----
    zero_f32<<<blocks((long long)N * Hid), T, 0, stream>>>(agg1, N * Hid);
    zero_f32<<<blocks(N), T, 0, stream>>>((float*)emaxu, N);   // key 0 == -inf
    zero_f32<<<blocks(N), T, 0, stream>>>(denom, N);

    wmma_gemm_bf16<Fin, Hid, false, false><<<dim3(tilesM, Hid/16), 32, 0, stream>>>(
        x, (const bpack32*)BpW1, nullptr, h0, N);
    attn_dots<Hid><<<blocks(N), T, 0, stream>>>(h0, a_src1, a_dst1, asv, adv, N);
    edge_max<<<blocks(EE), T, 0, stream>>>(src, dst, asv, adv, emaxu, EE, E);
    edge_exp<<<blocks(EE), T, 0, stream>>>(src, dst, asv, adv, emaxu, denom, eexp, EE, E);
    edge_agg<Hid><<<blocks((long long)EE * (Hid/4)), T, 0, stream>>>(
        src, dst, eexp, denom, h0, agg1, EE, E);
    bias_act<<<blocks((long long)N * Hid), T, 0, stream>>>(agg1, b1, agg1, N, Hid, 1);

    // ---- layer 2 ----
    zero_f32<<<blocks((long long)N * Out), T, 0, stream>>>(agg2, N * Out);
    zero_f32<<<blocks(N), T, 0, stream>>>((float*)emaxu, N);
    zero_f32<<<blocks(N), T, 0, stream>>>(denom, N);

    wmma_gemm_bf16<Hid, Out, false, false><<<dim3(tilesM, Out/16), 32, 0, stream>>>(
        agg1, (const bpack32*)BpW2, nullptr, h2, N);
    attn_dots<Out><<<blocks(N), T, 0, stream>>>(h2, a_src2, a_dst2, asv, adv, N);
    edge_max<<<blocks(EE), T, 0, stream>>>(src, dst, asv, adv, emaxu, EE, E);
    edge_exp<<<blocks(EE), T, 0, stream>>>(src, dst, asv, adv, emaxu, denom, eexp, EE, E);
    edge_agg<Out><<<blocks((long long)EE * (Out/4)), T, 0, stream>>>(
        src, dst, eexp, denom, h2, agg2, EE, E);
    bias_act<<<blocks((long long)N * Out), T, 0, stream>>>(agg2, b2, agg2, N, Out, 0);

    // ---- MLP head: relu(out2 @ lin1W + lin1b) @ lin2W + lin2b, softmax ----
    wmma_gemm_bf16<Out, Hid, true, true><<<dim3(tilesM, Hid/16), 32, 0, stream>>>(
        agg2, (const bpack32*)BpL1, lin1b, h0, N);    // K=16 zero-padded to 32
    wmma_gemm_bf16<Hid, Out, true, false><<<dim3(tilesM, Out/16), 32, 0, stream>>>(
        h0, (const bpack32*)BpL2, lin2b, outp, N);
    softmax16<<<blocks(N), T, 0, stream>>>(outp, N);
}
